// VoxelNet_1219770712576
// MI455X (gfx1250) — compile-verified, matrix operand order checked by reading, not run on hardware
//
#include <hip/hip_runtime.h>

#define BATCH 8
#define NANCH 107136
#define KPRE  2048
#define KPOST 128
#define CAND  (BATCH*KPRE)
#define IOU_TH 0.5f
#define SCORE_TH 0.05f
#define PI_F 3.14159265358979323846f

typedef __attribute__((ext_vector_type(16))) _Float16 v16h;
typedef __attribute__((ext_vector_type(8)))  float    v8f;

// candidate SoA field indices (float fields / u32 fields)
enum { F_X=0,F_Y,F_Z,F_W,F_L,F_H,F_R,F_X1,F_Y1,F_X2,F_Y2,F_AREA,F_SCORE,NFF };
enum { U_DIR=0,U_VALID,NUF };

// ---- CDNA5 data-mover feature detection (guarded so compile never regresses) --
#ifdef __has_builtin
# if __has_builtin(__builtin_amdgcn_tensor_load_to_lds)
#  define HAVE_TDM 1
# endif
# if __has_builtin(__builtin_amdgcn_global_load_async_to_lds_b32)
#  define HAVE_ASYNC_B32 1
# endif
# if __has_builtin(__builtin_amdgcn_s_wait_asynccnt)
#  define HAVE_WAIT_ASYNC 1
# endif
#endif
#ifndef HAVE_TDM
# define HAVE_TDM 0
#endif
#ifndef HAVE_ASYNC_B32
# define HAVE_ASYNC_B32 0
#endif
#ifndef HAVE_WAIT_ASYNC
# define HAVE_WAIT_ASYNC 0
#endif

#if HAVE_TDM
typedef __attribute__((ext_vector_type(4))) unsigned uint32x4;
typedef __attribute__((ext_vector_type(8))) int      int32x8;
typedef __attribute__((ext_vector_type(4))) int      int32x4;

__device__ __forceinline__ unsigned lds_byte_off(const void* p){
  return (unsigned)(unsigned long long)(__attribute__((address_space(3))) const void*)p;
}

// 2D TDM tile load: tile0 contiguous 4-byte elements, tile1 rows at stride0 elements.
__device__ __forceinline__ void tdm_load_2d(const void* gaddr, const void* lds,
                                            unsigned tile0, unsigned tile1,
                                            unsigned long long stride0){
  unsigned long long ga = (unsigned long long)gaddr;
  unsigned la = lds_byte_off(lds);
  // D# group0: count=1 (valid), lds_addr, global_addr[56:0], type=2 ("image")
  uint32x4 g0 = { 1u, la, (unsigned)ga,
                  (unsigned)((ga >> 32) & 0x01FFFFFFu) | (2u << 30) };
  // D# group1: data_size=2 (4B), tensor_dim0=tile0, tensor_dim1=tile1,
  //            tile_dim0=tile0, tile_dim1=tile1, tile_dim2=0, dim0 stride, dim1 stride=0
  int32x8 g1 = { (int)(2u << 16),
                 (int)((tile0 & 0xFFFFu) << 16),
                 (int)((tile0 >> 16) | ((tile1 & 0xFFFFu) << 16)),
                 (int)((tile1 >> 16) | ((tile0 & 0xFFFFu) << 16)),
                 (int)(tile1 & 0xFFFFu),
                 (int)(unsigned)(stride0 & 0xFFFFFFFFull),
                 (int)(unsigned)((stride0 >> 32) & 0xFFFFu),
                 0 };
  int32x4 z4 = { 0, 0, 0, 0 };
#if __clang_major__ >= 23
  int32x8 z8 = { 0, 0, 0, 0, 0, 0, 0, 0 };
  __builtin_amdgcn_tensor_load_to_lds(g0, g1, z4, z4, z8, 0);
#else
  __builtin_amdgcn_tensor_load_to_lds(g0, g1, z4, z4, 0);
#endif
}
#endif // HAVE_TDM

#if HAVE_ASYNC_B32
typedef __attribute__((address_space(1))) int gint_t;   // prints as "__device__ int"
typedef __attribute__((address_space(3))) int lint_t;   // prints as "__shared__ int"
__device__ __forceinline__ void async_g2l_b32(const void* g, void* l){
  __builtin_amdgcn_global_load_async_to_lds_b32((gint_t*)g, (lint_t*)l, 0, 0);
}
__device__ __forceinline__ void wait_async0(){
#if HAVE_WAIT_ASYNC
  __builtin_amdgcn_s_wait_asynccnt(0);
#else
  asm volatile("s_wait_asynccnt 0x0" ::: "memory");
#endif
}
#endif // HAVE_ASYNC_B32

__device__ __forceinline__ unsigned f2key(float f){
  unsigned u = __float_as_uint(f);
  return (u & 0x80000000u) ? ~u : (u | 0x80000000u);   // order-preserving float->uint
}
__device__ __forceinline__ float key2f(unsigned k){
  unsigned u = (k & 0x80000000u) ? (k ^ 0x80000000u) : ~k;
  return __uint_as_float(u);
}

// ---------------- Pass 1: scores = softmax(cls)[...,1], masked by threshold ----
__global__ void k_score(const float* __restrict__ cls, float* __restrict__ scores){
  int i = blockIdx.x*blockDim.x + threadIdx.x;
  if (i >= BATCH*NANCH) return;
  __builtin_prefetch(cls + (size_t)i*2 + 4096, 0, 1);
  float c0 = cls[(size_t)i*2+0], c1 = cls[(size_t)i*2+1];
  float s = 1.0f / (1.0f + __expf(c0 - c1));
  scores[i] = (s >= SCORE_TH) ? s : -__builtin_inff();
}

// ---------------- Pass 2: exact per-batch top-2048 (radix select + bitonic) ----
__global__ void __launch_bounds__(1024)
k_topk(const float* __restrict__ scores,
       unsigned* __restrict__ sel_key, unsigned* __restrict__ sel_idx){
  int b = blockIdx.x;
  const float* s = scores + (size_t)b*NANCH;
  __shared__ unsigned hist[256];
  __shared__ unsigned sh_prefix, sh_pmask, sh_krem;
  __shared__ unsigned cnt_gt, cnt_eq;
  __shared__ unsigned long long buf[KPRE];
  int t = threadIdx.x, nt = blockDim.x;

  unsigned prefix = 0, pmask = 0, krem = KPRE;
  for (int pass = 0; pass < 4; ++pass){
    int shift = 24 - pass*8;
    for (int i = t; i < 256; i += nt) hist[i] = 0u;
    __syncthreads();
    for (int i = t; i < NANCH; i += nt){
      unsigned key = f2key(s[i]);
      if ((key & pmask) == prefix) atomicAdd(&hist[(key >> shift) & 0xFFu], 1u);
    }
    __syncthreads();
    if (t == 0){
      unsigned cum = 0; int sel = 0; sh_krem = 1u;
      for (int bin = 255; bin >= 0; --bin){
        unsigned c = hist[bin];
        if (cum + c >= krem){ sel = bin; sh_krem = krem - cum; break; }
        cum += c;
      }
      sh_prefix = prefix | ((unsigned)sel << shift);
      sh_pmask  = pmask  | (0xFFu << shift);
    }
    __syncthreads();
    prefix = sh_prefix; pmask = sh_pmask; krem = sh_krem;
    __syncthreads();
  }
  unsigned T = prefix;                // exact 2048-th largest key
  if (t == 0){ cnt_gt = 0u; cnt_eq = 0u; }
  __syncthreads();
  unsigned base_eq = KPRE - krem;     // count of keys strictly > T
  for (int i = t; i < NANCH; i += nt){
    unsigned key = f2key(s[i]);
    if (key > T){
      unsigned p = atomicAdd(&cnt_gt, 1u);
      if (p < KPRE) buf[p] = ((unsigned long long)key << 32) | (unsigned)~(unsigned)i;
    } else if (key == T){
      unsigned e = atomicAdd(&cnt_eq, 1u);
      if (e < krem) buf[base_eq + e] = ((unsigned long long)key << 32) | (unsigned)~(unsigned)i;
    }
  }
  __syncthreads();
  // bitonic sort, descending by key, ties -> ascending index (idx stored as ~idx)
  for (unsigned k = 2; k <= KPRE; k <<= 1){
    for (unsigned j = k >> 1; j > 0; j >>= 1){
      for (unsigned i = t; i < KPRE; i += nt){
        unsigned ixj = i ^ j;
        if (ixj > i){
          unsigned long long a = buf[i], c = buf[ixj];
          bool desc = ((i & k) == 0u);
          if (desc ? (a < c) : (a > c)){ buf[i] = c; buf[ixj] = a; }
        }
      }
      __syncthreads();
    }
  }
  for (int i = t; i < KPRE; i += nt){
    unsigned long long v = buf[i];
    sel_key[(size_t)b*KPRE + i] = (unsigned)(v >> 32);
    sel_idx[(size_t)b*KPRE + i] = ~(unsigned)v;
  }
}

// ---------------- Pass 3: gather + decode + standup/area for selected 2048 ----
__global__ void k_decode(const float* __restrict__ box_preds, const float* __restrict__ anchors,
                         const float* __restrict__ dir_preds,
                         const unsigned* __restrict__ sel_key, const unsigned* __restrict__ sel_idx,
                         float* __restrict__ cf, unsigned* __restrict__ cu){
  int g = blockIdx.x*blockDim.x + threadIdx.x;
  if (g >= CAND) return;
  int b = g >> 11;
  unsigned idx = sel_idx[g];
  float sc = key2f(sel_key[g]);
  size_t base = (size_t)b*NANCH + idx;
  const float* bp = box_preds + base*7;
  const float* an = anchors   + base*7;
  const float* dp = dir_preds + base*2;
  float xa=an[0], ya=an[1], za=an[2], wa=an[3], la=an[4], ha=an[5], ra=an[6];
  float xt=bp[0], yt=bp[1], zt=bp[2], wt=bp[3], lt=bp[4], ht=bp[5], rt=bp[6];
  za += ha*0.5f;
  float diag = sqrtf(la*la + wa*wa);
  float xg = xt*diag + xa, yg = yt*diag + ya, zg = zt*ha + za;
  float lg = __expf(lt)*la, wg = __expf(wt)*wa, hg = __expf(ht)*ha;
  float rg = rt + ra;
  zg -= hg*0.5f;
  float cc = fabsf(cosf(rg)), ss = fabsf(sinf(rg));
  float hx = 0.5f*(wg*cc + lg*ss), hy = 0.5f*(wg*ss + lg*cc);
  cf[F_X*CAND+g]=xg;  cf[F_Y*CAND+g]=yg;  cf[F_Z*CAND+g]=zg;
  cf[F_W*CAND+g]=wg;  cf[F_L*CAND+g]=lg;  cf[F_H*CAND+g]=hg;  cf[F_R*CAND+g]=rg;
  cf[F_X1*CAND+g]=xg-hx; cf[F_Y1*CAND+g]=yg-hy;
  cf[F_X2*CAND+g]=xg+hx; cf[F_Y2*CAND+g]=yg+hy;
  cf[F_AREA*CAND+g]=(2.0f*hx)*(2.0f*hy);
  cf[F_SCORE*CAND+g]=sc;
  cu[U_DIR*CAND+g]   = (dp[1] > dp[0]) ? 1u : 0u;
  cu[U_VALID*CAND+g] = (sc >= SCORE_TH) ? 1u : 0u;
}

// ---------------- Pass 4: 2048x2048 IoU>thresh bitmask --------------------------
// One wave owns a 16-row x 2048-col stripe. The 2048x5 column-side SoA tile
// (X1,Y1,X2,Y2,AREA at stride CAND) is staged into LDS with one TDM descriptor;
// the 16-row side uses async global->LDS loads. WMMA computes the rank-2
// outer-sum area_i+area_j term of each 16x16 IoU tile's denominator.
__global__ void k_ioumask(const float* __restrict__ cf, unsigned* __restrict__ mm){
  int lane    = threadIdx.x;
  int rowbase = blockIdx.x * 16;      // 0..127 row tiles
  int b       = blockIdx.y;
  size_t gb = (size_t)b * KPRE;

  __shared__ float cxy[5][KPRE];      // col stripe: X1,Y1,X2,Y2,AREA  (40 KB)
  __shared__ float rxy[5][16];        // row tile

  const float* stripep = cf + (size_t)F_X1*CAND + gb;   // fields X1..AREA contiguous rows

#if HAVE_TDM
  tdm_load_2d(stripep, &cxy[0][0], KPRE, 5u, (unsigned long long)CAND);
#else
  for (int i = lane; i < 5*KPRE; i += 32){
    int f = i >> 11, c = i & (KPRE-1);
    cxy[f][c] = stripep[(size_t)f*CAND + c];
  }
#endif

#if HAVE_ASYNC_B32
  if (lane < 16){
    #pragma unroll
    for (int f = 0; f < 5; ++f)
      async_g2l_b32(&stripep[(size_t)f*CAND + rowbase + lane], &rxy[f][lane]);
  }
  wait_async0();
#else
  if (lane < 16){
    #pragma unroll
    for (int f = 0; f < 5; ++f)
      rxy[f][lane] = stripep[(size_t)f*CAND + rowbase + lane];
  }
#endif
#if HAVE_TDM
  __builtin_amdgcn_s_wait_tensorcnt(0);
#endif
  __syncthreads();

  // A (16x32 f16): A[m][0]=area_row[m], A[m][1]=1, rest 0.
  v16h amat = {};
  if (lane < 16){ amat[0] = (_Float16)rxy[4][lane]; amat[1] = (_Float16)1.0f; }

  int mhalf = (lane >> 4) << 3;       // D rows: lanes0-15 -> M=v, lanes16-31 -> M=v+8
  int nl = lane & 15;

  for (int colpair = 0; colpair < 64; ++colpair){
    unsigned rowbits[16];
    #pragma unroll
    for (int r = 0; r < 16; ++r) rowbits[r] = 0u;

    #pragma unroll
    for (int t = 0; t < 2; ++t){
      int cb = colpair*32 + t*16;
      // B (32x16 f16): B[0][n]=1, B[1][n]=area_col[n], rest 0.
      v16h bmat = {};
      if (lane < 16){ bmat[0] = (_Float16)1.0f; bmat[1] = (_Float16)cxy[4][cb + lane]; }
      v8f cacc = {};
      v8f d = __builtin_amdgcn_wmma_f32_16x16x32_f16(
          false, amat, false, bmat, (short)0, cacc, false, false); // d = area_m + area_n
      int nloc = cb + nl;
      float bx1 = cxy[0][nloc], by1 = cxy[1][nloc];
      float bx2 = cxy[2][nloc], by2 = cxy[3][nloc];
      #pragma unroll
      for (int v = 0; v < 8; ++v){
        int m = v + mhalf;
        float ix = fminf(rxy[2][m], bx2) - fmaxf(rxy[0][m], bx1); ix = fmaxf(ix, 0.0f);
        float iy = fminf(rxy[3][m], by2) - fmaxf(rxy[1][m], by1); iy = fmaxf(iy, 0.0f);
        float inter = ix * iy;
        float denom = fmaxf(d[v] - inter, 1e-8f);
        bool sup = inter > IOU_TH * denom;          // iou > 0.5
        unsigned bal = __builtin_amdgcn_ballot_w32(sup);
        rowbits[v]   |= (bal & 0xFFFFu)         << (16*t);
        rowbits[v+8] |= ((bal >> 16) & 0xFFFFu) << (16*t);
      }
    }
    if (lane == 0){
      #pragma unroll
      for (int r = 0; r < 16; ++r)
        mm[(gb + rowbase + r)*64 + colpair] = rowbits[r];
    }
  }
}

// ---------------- Pass 5: greedy NMS scan (1 wave/batch) + finalize ------------
__global__ void k_nms(const unsigned* __restrict__ mm, const float* __restrict__ cf,
                      const unsigned* __restrict__ cu, float* __restrict__ out){
  int b = blockIdx.x, lane = threadIdx.x;
  __shared__ unsigned rem[64];            // 2048-bit removed set
  __shared__ unsigned short kept[KPOST];
  __shared__ int kc;
  rem[lane] = 0u; rem[lane+32] = 0u;
  if (lane == 0) kc = 0;
  __syncthreads();
  size_t gb = (size_t)b * KPRE;
  const unsigned* mrow = mm + gb*64;

  for (int i = 0; i < KPRE; ++i){
    bool alive = ((rem[i>>5] >> (i & 31)) & 1u) == 0u;
    bool val   = cu[U_VALID*CAND + gb + i] != 0u;
    if (alive && val){
      if (lane == 0){ if (kc < KPOST) kept[kc] = (unsigned short)i; kc++; }
      unsigned w0 = mrow[(size_t)i*64 + lane];
      unsigned w1 = mrow[(size_t)i*64 + lane + 32];
      int iw = i >> 5; unsigned ib = (unsigned)(i & 31);
      unsigned tail = ~((2u << ib) - 1u);                 // bits j>i within word iw
      unsigned k0 = (lane    < iw) ? 0u : ((lane    == iw) ? tail : 0xFFFFFFFFu);
      unsigned k1 = (lane+32 < iw) ? 0u : ((lane+32 == iw) ? tail : 0xFFFFFFFFu);
      rem[lane]    |= w0 & k0;
      rem[lane+32] |= w1 & k1;
    }
    __syncthreads();
    if (kc >= KPOST) break;               // uniform
  }
  __syncthreads();
  int kcount = kc < KPOST ? kc : KPOST;

  float* outB = out;                          // [B][128][7]
  float* outS = out + (size_t)BATCH*KPOST*7;  // [B][128]
  float* outL = outS + BATCH*KPOST;           // labels (always 0)
  float* outM = outL + BATCH*KPOST;           // mask (0/1)
  for (int j = lane; j < KPOST; j += 32){
    float o0=0,o1=0,o2=0,o3=0,o4=0,o5=0,o6=0, os=0, om=0;
    if (j < kcount){
      size_t g = gb + kept[j];
      float x=cf[F_X*CAND+g], y=cf[F_Y*CAND+g], z=cf[F_Z*CAND+g];
      float w=cf[F_W*CAND+g], l=cf[F_L*CAND+g], h=cf[F_H*CAND+g];
      float r=cf[F_R*CAND+g];
      float dl = (float)cu[U_DIR*CAND+g];
      float dir_rot = r - floorf(r / PI_F) * PI_F;   // limit_period, offset 0, period pi
      float nr = dir_rot + PI_F * dl;
      bool inr = (x >= 0.0f) & (x <= 69.12f) &
                 (y >= -39.68f) & (y <= 39.68f) &
                 (z >= -5.0f) & (z <= 5.0f);
      if (inr){ o0=x;o1=y;o2=z;o3=w;o4=l;o5=h;o6=nr; os=cf[F_SCORE*CAND+g]; om=1.0f; }
    }
    float* pb = outB + ((size_t)b*KPOST + j)*7;
    pb[0]=o0; pb[1]=o1; pb[2]=o2; pb[3]=o3; pb[4]=o4; pb[5]=o5; pb[6]=o6;
    outS[b*KPOST+j]=os; outL[b*KPOST+j]=0.0f; outM[b*KPOST+j]=om;
  }
}

extern "C" void kernel_launch(void* const* d_in, const int* in_sizes, int n_in,
                              void* d_out, int out_size, void* d_ws, size_t ws_size,
                              hipStream_t stream) {
  (void)in_sizes; (void)n_in; (void)out_size; (void)ws_size;
  const float* box_preds = (const float*)d_in[0];
  const float* cls_preds = (const float*)d_in[1];
  const float* dir_preds = (const float*)d_in[2];
  const float* anchors   = (const float*)d_in[3];
  float* out = (float*)d_out;

  char* p = (char*)d_ws;
  auto carve = [&](size_t bytes)->char*{ char* r = p; p += (bytes + 255) & ~(size_t)255; return r; };
  float*    scores = (float*)   carve(sizeof(float)*(size_t)BATCH*NANCH);   // 3.43 MB
  unsigned* selk   = (unsigned*)carve(4u*(size_t)CAND);
  unsigned* seli   = (unsigned*)carve(4u*(size_t)CAND);
  float*    candF  = (float*)   carve(sizeof(float)*(size_t)NFF*CAND);      // 0.85 MB
  unsigned* candU  = (unsigned*)carve(4u*(size_t)NUF*CAND);
  unsigned* mm     = (unsigned*)carve(4u*(size_t)CAND*64);                  // 4 MB bitmask

  int total = BATCH*NANCH;
  k_score <<<(total + 255)/256, 256, 0, stream>>>(cls_preds, scores);
  k_topk  <<<BATCH, 1024, 0, stream>>>(scores, selk, seli);
  k_decode<<<(CAND + 255)/256, 256, 0, stream>>>(box_preds, anchors, dir_preds,
                                                 selk, seli, candF, candU);
  k_ioumask<<<dim3(128, BATCH), 32, 0, stream>>>(candF, mm);
  k_nms   <<<BATCH, 32, 0, stream>>>(mm, candF, candU, out);
}
